// VirConvAx_34351148434346
// MI455X (gfx1250) — matrix operand order repaired and checked by reading, not compiled, not run on hardware
//
#include <hip/hip_runtime.h>
#include <hip/hip_bf16.h>
#include <math.h>

// ---------------------------------------------------------------------------
// MI455X (gfx1250) implementation: wave32 WMMA f32_16x16x32_f16 GEMM tiles.
// Multi-pass because batch-norm statistics are over the full 1M-row batch.
// All intermediates stored f16 in d_ws (bandwidth-bound problem).
// ---------------------------------------------------------------------------

typedef __attribute__((ext_vector_type(16))) _Float16 v16h;
typedef __attribute__((ext_vector_type(8)))  _Float16 v8h;
typedef __attribute__((ext_vector_type(8)))  float    v8f;

#define GEMM_BLOCKS  1024
#define GEMM_THREADS 256
#define WAVES_PB     (GEMM_THREADS / 32)
#define EW_BLOCKS    2048
#define EW_THREADS   256

static __device__ inline v8f wmma_f16(v16h a, v16h b, v8f c) {
    // emits v_wmma_f32_16x16x32_f16
    return __builtin_amdgcn_wmma_f32_16x16x32_f16(
        /*neg_a=*/false, a, /*neg_b=*/false, b,
        /*c_mod=*/(short)0, c, /*reuse_a=*/false, /*reuse_b=*/false);
}

// B fragment (32x16, f16) from f32 row-major weights W[n][k], ld = ldw.
// lane: n = 16*nb + (lane&15); element e -> K = kbase + e (kbase = kb*32 + 16*h)
static __device__ inline v16h load_bfrag_f32(const float* W, int ldw, int n, int kbase) {
    v16h b;
    const float4* p = (const float4*)(W + (size_t)n * ldw + kbase);
    float4 f0 = p[0], f1 = p[1], f2 = p[2], f3 = p[3];
    b[0]=(_Float16)f0.x;  b[1]=(_Float16)f0.y;  b[2]=(_Float16)f0.z;  b[3]=(_Float16)f0.w;
    b[4]=(_Float16)f1.x;  b[5]=(_Float16)f1.y;  b[6]=(_Float16)f1.z;  b[7]=(_Float16)f1.w;
    b[8]=(_Float16)f2.x;  b[9]=(_Float16)f2.y;  b[10]=(_Float16)f2.z; b[11]=(_Float16)f2.w;
    b[12]=(_Float16)f3.x; b[13]=(_Float16)f3.y; b[14]=(_Float16)f3.z; b[15]=(_Float16)f3.w;
    return b;
}

// Gate B-tile: column 0 = folded vector g0, column 1 = g1, rest zero.
static __device__ inline v16h load_gfrag(const float* g0, const float* g1, int nl, int kbase) {
    v16h b;
#pragma unroll
    for (int e = 0; e < 16; ++e) b[e] = (_Float16)0.0f;
    if (nl == 0) {
#pragma unroll
        for (int e = 0; e < 16; ++e) b[e] = (_Float16)g0[kbase + e];
    } else if (nl == 1) {
#pragma unroll
        for (int e = 0; e < 16; ++e) b[e] = (_Float16)g1[kbase + e];
    }
    return b;
}

// A fragment (16x32, f16) from f32 row (layout: element e -> K=koff+e)
static __device__ inline v16h load_afrag_f32(const float* rowp, int koff) {
    v16h a;
    const float4* c0 = (const float4*)(rowp + koff);
    const float4* c1 = (const float4*)(rowp + koff + 8);
    float4 f0 = c0[0], f1 = c0[1], f2 = c1[0], f3 = c1[1];
    a[0]=(_Float16)f0.x;  a[1]=(_Float16)f0.y;  a[2]=(_Float16)f0.z;  a[3]=(_Float16)f0.w;
    a[4]=(_Float16)f1.x;  a[5]=(_Float16)f1.y;  a[6]=(_Float16)f1.z;  a[7]=(_Float16)f1.w;
    a[8]=(_Float16)f2.x;  a[9]=(_Float16)f2.y;  a[10]=(_Float16)f2.z; a[11]=(_Float16)f2.w;
    a[12]=(_Float16)f3.x; a[13]=(_Float16)f3.y; a[14]=(_Float16)f3.z; a[15]=(_Float16)f3.w;
    return a;
}

static __device__ inline v16h load_afrag_f16(const _Float16* rowp, int koff) {
    v16h a;
    v8h u = *(const v8h*)(rowp + koff);
    v8h v = *(const v8h*)(rowp + koff + 8);
#pragma unroll
    for (int j = 0; j < 8; ++j) { a[j] = u[j]; a[8 + j] = v[j]; }
    return a;
}

// f16 load + batchnorm(scale,shift) + relu, back to f16 for the next WMMA.
static __device__ inline v16h load_afrag_bnrelu(const _Float16* rowp, int koff,
                                                const float* sc0, const float* sh0,
                                                const float* sc1, const float* sh1) {
    v16h a;
    v8h u = *(const v8h*)(rowp + koff);
    v8h v = *(const v8h*)(rowp + koff + 8);
#pragma unroll
    for (int j = 0; j < 8; ++j) {
        float f = (float)u[j]; f = fmaxf(f * sc0[j] + sh0[j], 0.0f); a[j]     = (_Float16)f;
        float g = (float)v[j]; g = fmaxf(g * sc1[j] + sh1[j], 0.0f); a[8 + j] = (_Float16)g;
    }
    return a;
}

// ---------------------------------------------------------------------------
// 64x64 GEMM pass. MODE 0: f32 in, write gate partial dot.
//                  MODE 1: f32 in, finish gate logit + sigmoid -> weights.
//                  MODE 2: f16 in, apply bn(scale/shift)+relu to input first.
// Always: z_out = x' @ W^T + bias (f16), per-channel sum/sumsq partials.
// ---------------------------------------------------------------------------
template <int MODE>
__global__ void __launch_bounds__(GEMM_THREADS)
gemm64_kernel(const void* __restrict__ xin, const float* __restrict__ W,
              const float* __restrict__ bias, const float* __restrict__ scsh_in,
              const float* __restrict__ gcol0, const float* __restrict__ gcol1,
              const float* __restrict__ gconst, float* __restrict__ gatebuf,
              _Float16* __restrict__ zout, float* __restrict__ stat_partial, int N) {
    __shared__ float red[WAVES_PB][128];
    const int lane = threadIdx.x & 31;
    const int wave = threadIdx.x >> 5;
    const int h = lane >> 4;
    const int nl = lane & 15;

    v16h bfrag[2][4];
#pragma unroll
    for (int kb = 0; kb < 2; ++kb)
#pragma unroll
        for (int nb = 0; nb < 4; ++nb)
            bfrag[kb][nb] = load_bfrag_f32(W, 64, nb * 16 + nl, kb * 32 + 16 * h);

    v16h gfrag[2];
    if (MODE < 2) {
#pragma unroll
        for (int kb = 0; kb < 2; ++kb)
            gfrag[kb] = load_gfrag(gcol0, gcol1, nl, kb * 32 + 16 * h);
    }

    float bia[4];
#pragma unroll
    for (int nb = 0; nb < 4; ++nb) bia[nb] = bias[nb * 16 + nl];

    float scr[4][8], shr[4][8];
    if (MODE == 2) {
#pragma unroll
        for (int m = 0; m < 4; ++m)
#pragma unroll
            for (int j = 0; j < 8; ++j) {
                int c = 16 * h + 32 * (m >> 1) + 8 * (m & 1) + j;
                scr[m][j] = scsh_in[c];
                shr[m][j] = scsh_in[64 + c];
            }
    }

    float sums[4] = {0.f, 0.f, 0.f, 0.f}, ssqs[4] = {0.f, 0.f, 0.f, 0.f};
    const int nt = (N + 15) >> 4;
    for (int t = blockIdx.x * WAVES_PB + wave; t < nt; t += gridDim.x * WAVES_PB) {
        const int row0 = t << 4;
        const bool full = (row0 + 16 <= N);   // wave-uniform fast-path predicate
        int ar = row0 + nl;
        if (ar >= N) ar = N - 1;

        v16h afrag[2];
        if (MODE == 2) {
            const _Float16* rp = (const _Float16*)xin + (size_t)ar * 64;
#pragma unroll
            for (int kb = 0; kb < 2; ++kb)
                afrag[kb] = load_afrag_bnrelu(rp, kb * 32 + 16 * h,
                                              scr[2 * kb], shr[2 * kb],
                                              scr[2 * kb + 1], shr[2 * kb + 1]);
        } else {
            const float* rp = (const float*)xin + (size_t)ar * 64;
#pragma unroll
            for (int kb = 0; kb < 2; ++kb)
                afrag[kb] = load_afrag_f32(rp, kb * 32 + 16 * h);
        }

        // bias folded into the accumulator (WMMA C operand)
        v8f acc[4], gac;
#pragma unroll
        for (int nb = 0; nb < 4; ++nb)
#pragma unroll
            for (int r = 0; r < 8; ++r) acc[nb][r] = bia[nb];
#pragma unroll
        for (int r = 0; r < 8; ++r) gac[r] = 0.0f;

#pragma unroll
        for (int kb = 0; kb < 2; ++kb) {
#pragma unroll
            for (int nb = 0; nb < 4; ++nb)
                acc[nb] = wmma_f16(afrag[kb], bfrag[kb][nb], acc[nb]);
            if (MODE < 2) gac = wmma_f16(afrag[kb], gfrag[kb], gac);
        }

        _Float16* zp = zout + (size_t)(row0 + 8 * h) * 64 + nl;
        if (full) {
#pragma unroll
            for (int nb = 0; nb < 4; ++nb)
#pragma unroll
                for (int r = 0; r < 8; ++r) {
                    float v = acc[nb][r];
                    sums[nb] += v;
                    ssqs[nb] += v * v;
                    zp[(size_t)r * 64 + nb * 16] = (_Float16)v;
                }
        } else {
#pragma unroll
            for (int nb = 0; nb < 4; ++nb)
#pragma unroll
                for (int r = 0; r < 8; ++r) {
                    int row = row0 + r + 8 * h;
                    float v = acc[nb][r];
                    if (row < N) {
                        sums[nb] += v;
                        ssqs[nb] += v * v;
                        zout[(size_t)row * 64 + (nb * 16 + nl)] = (_Float16)v;
                    }
                }
        }

        if (MODE == 0) {
            if (nl < 2) {
                if (full) {
#pragma unroll
                    for (int r = 0; r < 8; ++r)
                        gatebuf[(size_t)(row0 + r + 8 * h) * 2 + nl] = gac[r];
                } else {
#pragma unroll
                    for (int r = 0; r < 8; ++r) {
                        int row = row0 + r + 8 * h;
                        if (row < N) gatebuf[(size_t)row * 2 + nl] = gac[r];
                    }
                }
            }
        } else if (MODE == 1) {
            if (nl < 2) {
#pragma unroll
                for (int r = 0; r < 8; ++r) {
                    int row = row0 + r + 8 * h;
                    if (full || row < N) {
                        float lg = gatebuf[(size_t)row * 2 + nl] + gac[r] + gconst[nl];
                        gatebuf[(size_t)row * 2 + nl] = 1.0f / (1.0f + __expf(-lg));
                    }
                }
            }
        }
    }

    // deterministic reduction: lane-pair -> per-wave LDS slot -> ordered sum
#pragma unroll
    for (int nb = 0; nb < 4; ++nb) {
        sums[nb] += __shfl_xor(sums[nb], 16);
        ssqs[nb] += __shfl_xor(ssqs[nb], 16);
    }
    if (lane < 16) {
#pragma unroll
        for (int nb = 0; nb < 4; ++nb) {
            red[wave][nb * 16 + nl]      = sums[nb];
            red[wave][64 + nb * 16 + nl] = ssqs[nb];
        }
    }
    __syncthreads();
    if (threadIdx.x < 128) {
        float s = 0.f;
        for (int w = 0; w < WAVES_PB; ++w) s += red[w][threadIdx.x];
        stat_partial[(size_t)blockIdx.x * 128 + threadIdx.x] = s;
    }
}

// ---------------------------------------------------------------------------
// Fusion GEMM: z4 = [valid_att | pseudo_att](Nx128) @ fc4_w^T(64x128) + b
// ---------------------------------------------------------------------------
__global__ void __launch_bounds__(GEMM_THREADS)
gemm128_kernel(const _Float16* __restrict__ vatt, const _Float16* __restrict__ patt,
               const float* __restrict__ W, const float* __restrict__ bias,
               _Float16* __restrict__ zout, float* __restrict__ stat_partial, int N) {
    __shared__ float red[WAVES_PB][128];
    const int lane = threadIdx.x & 31;
    const int wave = threadIdx.x >> 5;
    const int h = lane >> 4;
    const int nl = lane & 15;

    v16h bfrag[4][4];
#pragma unroll
    for (int kb = 0; kb < 4; ++kb)
#pragma unroll
        for (int nb = 0; nb < 4; ++nb)
            bfrag[kb][nb] = load_bfrag_f32(W, 128, nb * 16 + nl, kb * 32 + 16 * h);

    float bia[4];
#pragma unroll
    for (int nb = 0; nb < 4; ++nb) bia[nb] = bias[nb * 16 + nl];

    float sums[4] = {0.f, 0.f, 0.f, 0.f}, ssqs[4] = {0.f, 0.f, 0.f, 0.f};
    const int nt = (N + 15) >> 4;
    for (int t = blockIdx.x * WAVES_PB + wave; t < nt; t += gridDim.x * WAVES_PB) {
        const int row0 = t << 4;
        const bool full = (row0 + 16 <= N);
        int ar = row0 + nl;
        if (ar >= N) ar = N - 1;

        // hoist all A-fragment loads (8 x b128) so they issue as one clause
        v16h afrag[4];
#pragma unroll
        for (int kb = 0; kb < 4; ++kb) {
            const _Float16* src = (kb < 2) ? vatt : patt;
            afrag[kb] = load_afrag_f16(src + (size_t)ar * 64, (kb & 1) * 32 + 16 * h);
        }

        v8f acc[4];
#pragma unroll
        for (int nb = 0; nb < 4; ++nb)
#pragma unroll
            for (int r = 0; r < 8; ++r) acc[nb][r] = bia[nb];

#pragma unroll
        for (int kb = 0; kb < 4; ++kb)
#pragma unroll
            for (int nb = 0; nb < 4; ++nb)
                acc[nb] = wmma_f16(afrag[kb], bfrag[kb][nb], acc[nb]);

        _Float16* zp = zout + (size_t)(row0 + 8 * h) * 64 + nl;
        if (full) {
#pragma unroll
            for (int nb = 0; nb < 4; ++nb)
#pragma unroll
                for (int r = 0; r < 8; ++r) {
                    float v = acc[nb][r];
                    sums[nb] += v;
                    ssqs[nb] += v * v;
                    zp[(size_t)r * 64 + nb * 16] = (_Float16)v;
                }
        } else {
#pragma unroll
            for (int nb = 0; nb < 4; ++nb)
#pragma unroll
                for (int r = 0; r < 8; ++r) {
                    int row = row0 + r + 8 * h;
                    float v = acc[nb][r];
                    if (row < N) {
                        sums[nb] += v;
                        ssqs[nb] += v * v;
                        zout[(size_t)row * 64 + (nb * 16 + nl)] = (_Float16)v;
                    }
                }
        }
    }

#pragma unroll
    for (int nb = 0; nb < 4; ++nb) {
        sums[nb] += __shfl_xor(sums[nb], 16);
        ssqs[nb] += __shfl_xor(ssqs[nb], 16);
    }
    if (lane < 16) {
#pragma unroll
        for (int nb = 0; nb < 4; ++nb) {
            red[wave][nb * 16 + nl]      = sums[nb];
            red[wave][64 + nb * 16 + nl] = ssqs[nb];
        }
    }
    __syncthreads();
    if (threadIdx.x < 128) {
        float s = 0.f;
        for (int w = 0; w < WAVES_PB; ++w) s += red[w][threadIdx.x];
        stat_partial[(size_t)blockIdx.x * 128 + threadIdx.x] = s;
    }
}

// sum/sumsq partials -> per-channel (scale, shift) for BN
__global__ void finalize_stats_kernel(const float* __restrict__ partial, int nblocks,
                                      const float* __restrict__ g, const float* __restrict__ b,
                                      float invN, float* __restrict__ scsh) {
    int c = threadIdx.x;
    if (c < 64) {
        float s = 0.f, q = 0.f;
        for (int blk = 0; blk < nblocks; ++blk) {
            s += partial[(size_t)blk * 128 + c];
            q += partial[(size_t)blk * 128 + 64 + c];
        }
        float m = s * invN;
        float var = q * invN - m * m;
        float sc = g[c] * rsqrtf(var + 1e-5f);
        scsh[c] = sc;
        scsh[64 + c] = b[c] - m * sc;
    }
}

// in-place: z = relu(z*sc + sh) * gate_weight[row]
__global__ void apply_att_kernel(_Float16* __restrict__ z, const float* __restrict__ scsh,
                                 const float* __restrict__ wbuf, int sel, long n8) {
    for (long i = (long)blockIdx.x * blockDim.x + threadIdx.x; i < n8;
         i += (long)gridDim.x * blockDim.x) {
        long base = i * 8;
        long row = base >> 6;
        int c0 = (int)(base & 63);
        float wgt = wbuf[row * 2 + sel];
        v8h v = *(v8h*)(z + base);
#pragma unroll
        for (int j = 0; j < 8; ++j) {
            int c = c0 + j;
            float f = (float)v[j];
            f = fmaxf(f * scsh[c] + scsh[64 + c], 0.0f) * wgt;
            v[j] = (_Float16)f;
        }
        *(v8h*)(z + base) = v;
    }
}

// out(f32) = relu(z4*sc + sh)
__global__ void final_out_kernel(const _Float16* __restrict__ z, const float* __restrict__ scsh,
                                 float* __restrict__ out, long n8) {
    for (long i = (long)blockIdx.x * blockDim.x + threadIdx.x; i < n8;
         i += (long)gridDim.x * blockDim.x) {
        long base = i * 8;
        int c0 = (int)(base & 63);
        v8h v = *(const v8h*)(z + base);
        float o[8];
#pragma unroll
        for (int j = 0; j < 8; ++j) {
            int c = c0 + j;
            o[j] = fmaxf((float)v[j] * scsh[c] + scsh[64 + c], 0.0f);
        }
        float4 o0 = {o[0], o[1], o[2], o[3]};
        float4 o1 = {o[4], o[5], o[6], o[7]};
        *(float4*)(out + base) = o0;
        *(float4*)(out + base + 4) = o1;
    }
}

// Fold fc1/fc2/fc3 into two 64-d dot vectors per sigmoid output + constants.
// gv layout: gA0[64], gA1[64], gB0[64], gB1[64], c0, c1
__global__ void gate_prep_kernel(const float* __restrict__ fc1w, const float* __restrict__ fc1b,
                                 const float* __restrict__ fc2w, const float* __restrict__ fc2b,
                                 const float* __restrict__ fc3w, const float* __restrict__ fc3b,
                                 float* __restrict__ gv) {
    int k = threadIdx.x;
    if (k < 64) {
        float a0 = 0.f, a1 = 0.f, b0 = 0.f, b1 = 0.f;
        for (int j = 0; j < 16; ++j) {
            float w1 = fc1w[j * 64 + k], w2 = fc2w[j * 64 + k];
            a0 += w1 * fc3w[j];
            a1 += w1 * fc3w[32 + j];
            b0 += w2 * fc3w[16 + j];
            b1 += w2 * fc3w[48 + j];
        }
        gv[k] = a0; gv[64 + k] = a1; gv[128 + k] = b0; gv[192 + k] = b1;
    }
    if (k < 2) {
        float c = fc3b[k];
        for (int j = 0; j < 16; ++j)
            c += fc1b[j] * fc3w[k * 32 + j] + fc2b[j] * fc3w[k * 32 + 16 + j];
        gv[256 + k] = c;
    }
}

extern "C" void kernel_launch(void* const* d_in, const int* in_sizes, int n_in,
                              void* d_out, int out_size, void* d_ws, size_t ws_size,
                              hipStream_t stream) {
    const float* pseudo = (const float*)d_in[0];
    const float* valid  = (const float*)d_in[1];
    const float* fc1w = (const float*)d_in[2];
    const float* fc1b = (const float*)d_in[3];
    const float* fc2w = (const float*)d_in[4];
    const float* fc2b = (const float*)d_in[5];
    const float* fc3w = (const float*)d_in[6];
    const float* fc3b = (const float*)d_in[7];
    const float* fc4w = (const float*)d_in[8];
    const float* fc4b = (const float*)d_in[9];
    const float* c1w  = (const float*)d_in[10];
    const float* c1b  = (const float*)d_in[11];
    const float* bnc1g = (const float*)d_in[12];
    const float* bnc1b = (const float*)d_in[13];
    const float* c2w  = (const float*)d_in[14];
    const float* c2b  = (const float*)d_in[15];
    const float* bnc2g = (const float*)d_in[16];
    const float* bnc2b = (const float*)d_in[17];
    const float* bn1g = (const float*)d_in[18];
    const float* bn1b = (const float*)d_in[19];

    const int N = in_sizes[0] / 64;
    float* out = (float*)d_out;

    char* ws = (char*)d_ws;
    size_t zbytes = (size_t)N * 64 * sizeof(_Float16);
    _Float16* bufA = (_Float16*)ws;                  // z1, later z4
    _Float16* bufB = (_Float16*)(ws + zbytes);       // z3, later valid_att (in place)
    _Float16* bufC = (_Float16*)(ws + 2 * zbytes);   // z2, later pseudo_att (in place)
    float* wbuf = (float*)(ws + 3 * zbytes);         // N x 2 gate partials -> sigmoid weights
    float* gv    = wbuf + (size_t)N * 2;             // 258 floats (folded gate vectors)
    float* scsh1 = gv + 264;
    float* scsh2 = scsh1 + 128;
    float* scsh3 = scsh2 + 128;
    float* scsh4 = scsh3 + 128;
    float* partial = scsh4 + 128;                    // GEMM_BLOCKS x 128

    const float invN = 1.0f / (float)N;
    dim3 gb(GEMM_BLOCKS), gt(GEMM_THREADS);
    long n8 = (long)N * 8;

    gate_prep_kernel<<<1, 64, 0, stream>>>(fc1w, fc1b, fc2w, fc2b, fc3w, fc3b, gv);

    // P1a: z1 = pseudo @ conv1^T + b; stats; gate partial (pseudo side)
    gemm64_kernel<0><<<gb, gt, 0, stream>>>(pseudo, c1w, c1b, nullptr,
                                            gv, gv + 64, nullptr, wbuf, bufA, partial, N);
    finalize_stats_kernel<<<1, 64, 0, stream>>>(partial, GEMM_BLOCKS, bnc1g, bnc1b, invN, scsh1);

    // P1b: z3 = valid @ conv2^T + b; stats; finish sigmoid gate weights
    gemm64_kernel<1><<<gb, gt, 0, stream>>>(valid, c2w, c2b, nullptr,
                                            gv + 128, gv + 192, gv + 256, wbuf, bufB, partial, N);
    finalize_stats_kernel<<<1, 64, 0, stream>>>(partial, GEMM_BLOCKS, bnc2g, bnc2b, invN, scsh3);

    // P2: pseudo' = relu(bn(z1)); z2 = pseudo' @ conv1^T + b; stats
    gemm64_kernel<2><<<gb, gt, 0, stream>>>(bufA, c1w, c1b, scsh1,
                                            nullptr, nullptr, nullptr, nullptr, bufC, partial, N);
    finalize_stats_kernel<<<1, 64, 0, stream>>>(partial, GEMM_BLOCKS, bnc1g, bnc1b, invN, scsh2);

    // valid_att / pseudo_att (in-place, gated)
    apply_att_kernel<<<EW_BLOCKS, EW_THREADS, 0, stream>>>(bufB, scsh3, wbuf, 1, n8);
    apply_att_kernel<<<EW_BLOCKS, EW_THREADS, 0, stream>>>(bufC, scsh2, wbuf, 0, n8);

    // P3: z4 = [valid_att | pseudo_att] @ fc4^T + b; stats
    gemm128_kernel<<<gb, gt, 0, stream>>>(bufB, bufC, fc4w, fc4b, bufA, partial, N);
    finalize_stats_kernel<<<1, 64, 0, stream>>>(partial, GEMM_BLOCKS, bn1g, bn1b, invN, scsh4);

    // P4: out = relu(bn(z4)) in f32
    final_out_kernel<<<EW_BLOCKS, EW_THREADS, 0, stream>>>(bufA, scsh4, out, n8);
}
